// RNN_14602888806402
// MI455X (gfx1250) — compile-verified, matrix operand order checked by reading, not run on hardware
//
#include <hip/hip_runtime.h>
#include <cstddef>

// ---------------------------------------------------------------------------
// Types for CDNA5 WMMA (wave32): V_WMMA_F32_16X16X32_BF16
// ---------------------------------------------------------------------------
typedef __attribute__((ext_vector_type(8)))  __bf16 v8bf;    // 16 bytes
typedef __attribute__((ext_vector_type(16))) __bf16 v16bf;   // 32 bytes = 8 VGPRs
typedef __attribute__((ext_vector_type(8)))  float  v8f;     // 8 VGPRs
typedef __attribute__((ext_vector_type(4)))  int    v4i;     // 16 bytes

union BF16x16 {
    v16bf v;
    v8bf  h[2];
};

// Per the ISA 16-bit A/B fragment layout (16x32, MxK):
//  lanes 0-15  : row = lane,    elements 0..7 = K=k0..k0+7,   8..15 = K=k0+16..k0+23
//  lanes 16-31 : row = lane-16, elements 0..7 = K=k0+8..k0+15, 8..15 = K=k0+24..k0+31
__device__ __forceinline__ BF16x16 load_frag(const __bf16* rowPtr, int kc, int koff) {
    BF16x16 f;
    f.h[0] = *(const v8bf*)(rowPtr + kc + koff);
    f.h[1] = *(const v8bf*)(rowPtr + kc + koff + 16);
    return f;
}

__device__ __forceinline__ v8f wmma_bf16(const BF16x16& a, const BF16x16& b, v8f c) {
    return __builtin_amdgcn_wmma_f32_16x16x32_bf16(
        /*neg_a=*/false, a.v, /*neg_b=*/false, b.v,
        /*c_mod=*/(short)0, c, /*reuse_a=*/false, /*reuse_b=*/false);
}

// gfx1250 async global->LDS copy (ASYNCcnt-tracked), guarded so the file
// still compiles on toolchains that don't expose the builtins.
// Probe-derived signature: (AS1 v4i* gsrc, AS3 v4i* ldst, imm offset, imm cpol)
#if __has_builtin(__builtin_amdgcn_global_load_async_to_lds_b128) && \
    __has_builtin(__builtin_amdgcn_s_wait_asynccnt)
#define USE_ASYNC_LDS 1
typedef __attribute__((address_space(1))) v4i*  as1_v4i_ptr;   // global src
typedef __attribute__((address_space(3))) void* as3_void_ptr;
typedef __attribute__((address_space(3))) v4i*  as3_v4i_ptr;   // LDS dst

__device__ __forceinline__ void async_copy_b128(const float* gsrc, float* lds_dst) {
    __builtin_amdgcn_global_load_async_to_lds_b128(
        (as1_v4i_ptr)(void*)(float*)gsrc,
        (as3_v4i_ptr)(as3_void_ptr)lds_dst, 0, 0);
}
#endif

// ---------------------------------------------------------------------------
// Problem constants
// ---------------------------------------------------------------------------
#define RNN_B   32
#define RNN_L   1024
#define RNN_H   512
#define RNN_M   (RNN_B * RNN_L)       // 32768 rows for the big GEMM

// ---------------------------------------------------------------------------
// Elementwise convert f32 -> bf16 (grid-stride)
// ---------------------------------------------------------------------------
__global__ void cvt_f32_to_bf16(const float* __restrict__ in, __bf16* __restrict__ out, int n) {
    for (int i = blockIdx.x * blockDim.x + threadIdx.x; i < n; i += gridDim.x * blockDim.x)
        out[i] = (__bf16)in[i];
}

// Transpose + convert: W [K x N] f32 (row-major) -> Wt [N x K] bf16 (row-major)
__global__ void transpose_to_bf16(const float* __restrict__ in, __bf16* __restrict__ out,
                                  int K, int N) {
    int idx = blockIdx.x * blockDim.x + threadIdx.x;
    if (idx < K * N) {
        int n = idx % N;
        int k = idx / N;
        out[n * K + k] = (__bf16)in[k * N + n];
    }
}

// bias[n] = BI[n] + BH[n]
__global__ void bias_sum(const float* __restrict__ bi, const float* __restrict__ bh,
                         float* __restrict__ out, int n) {
    int i = blockIdx.x * blockDim.x + threadIdx.x;
    if (i < n) out[i] = bi[i] + bh[i];
}

// ---------------------------------------------------------------------------
// Big GEMM: C[M x 512] = A[M x 512](bf16) @ Wt^T + bias   (Wt is [N x K] bf16)
// Block = 256 threads (8 waves); block tile = 128(M) x 64(N); wave = 16 x 64.
// ---------------------------------------------------------------------------
__global__ __launch_bounds__(256) void gemm_xw_bf16(
    const __bf16* __restrict__ A,    // [M x 512] row-major
    const __bf16* __restrict__ Wt,   // [512(N) x 512(K)] row-major (transposed weight)
    const float*  __restrict__ bias, // [512]
    float*        __restrict__ C,    // [M x 512]
    int M)
{
    const int K = RNN_H, N = RNN_H;
    const int lane  = threadIdx.x & 31;
    const int wave  = threadIdx.x >> 5;
    const int mTile = blockIdx.x * 8 + wave;   // 16 rows per wave
    const int nBase = blockIdx.y * 64;         // 4 tiles of 16 cols
    const int mRow  = mTile * 16 + (lane & 15);
    const int koff  = (lane >> 4) * 8;
    (void)M;

    const __bf16* aRow = A + (size_t)mRow * K;

    v8f acc[4] = {v8f{}, v8f{}, v8f{}, v8f{}};

    for (int kc = 0; kc < K; kc += 32) {
        if (kc + 64 < K)  // warm caches for the next A chunk (global_prefetch_b8)
            __builtin_prefetch((const void*)(aRow + kc + 64), 0, 3);
        BF16x16 a = load_frag(aRow, kc, koff);
#pragma unroll
        for (int j = 0; j < 4; ++j) {
            const __bf16* bRow = Wt + (size_t)(nBase + j * 16 + (lane & 15)) * K;
            BF16x16 b = load_frag(bRow, kc, koff);
            acc[j] = wmma_bf16(a, b, acc[j]);
        }
    }

    // C/D layout: VGPR v holds row (v + 8*(lane>>4)), col (lane&15) of the tile.
    const int mBase2 = mTile * 16 + 8 * (lane >> 4);
#pragma unroll
    for (int j = 0; j < 4; ++j) {
        const int n  = nBase + j * 16 + (lane & 15);
        const float bv = bias[n];
#pragma unroll
        for (int v = 0; v < 8; ++v)
            C[(size_t)(mBase2 + v) * N + n] = acc[j][v] + bv;
    }
}

// ---------------------------------------------------------------------------
// Recurrent scan: one workgroup per 16-row batch half (grid.x = 2).
// 1024 threads = 32 waves; wave w owns N-tile w. h half [16 x 512] bf16 in LDS.
// WH K-chunks 0..7 live in VGPRs for the whole scan; chunks 8..15 stream (L2).
// xw[t+1] slice is async-copied into an LDS double buffer one step ahead.
// ---------------------------------------------------------------------------
__global__ __launch_bounds__(1024, 1) void rnn_scan(
    const float*  __restrict__ xw,      // [B*L x 512] f32 (bias already folded in)
    const __bf16* __restrict__ WHt,     // [512(N) x 512(K)] bf16 transposed
    const float*  __restrict__ h0,      // [B x 512] f32 initial hidden (this layer)
    __bf16*       __restrict__ seq_bf,  // bf16 seq out (next layer input) or nullptr
    float*        __restrict__ seq_f32, // f32 seq out (final layer -> d_out) or nullptr
    float*        __restrict__ hN)      // [B x 512] f32 final hidden (this layer)
{
    __shared__ __bf16 hsm[16 * RNN_H];        // 16 KB hidden state
#if USE_ASYNC_LDS
    __shared__ float  xbuf[2][16 * RNN_H];    // 2 x 32 KB xw staging
#endif

    const int K    = RNN_H;
    const int tid  = threadIdx.x;
    const int lane = tid & 31;
    const int wave = tid >> 5;
    const int bOff = blockIdx.x * 16;   // batch rows [bOff, bOff+16)

    // load initial hidden state for our 16 batch rows
    for (int i = tid; i < 16 * RNN_H; i += 1024)
        hsm[i] = (__bf16)h0[(size_t)bOff * RNN_H + i];
    __syncthreads();

    const int nTile = wave;
    const int n     = nTile * 16 + (lane & 15);
    const int koff  = (lane >> 4) * 8;
    const int mA    = lane & 15;
    const int mHalf = 8 * (lane >> 4);

    const __bf16* bRow = WHt + (size_t)n * K;   // WH column n
    const __bf16* aRow = hsm + (size_t)mA * K;  // hidden row mA in LDS

    // --- register-resident half of WH (K = 0..255): 64 VGPRs, loop-invariant
    BF16x16 bReg[8];
#pragma unroll
    for (int c = 0; c < 8; ++c)
        bReg[c] = load_frag(bRow, c * 32, koff);

#if USE_ASYNC_LDS
    // prime the pipeline: stage xw[t=0] into buffer 0 (2 b128 chunks / thread)
#pragma unroll
    for (int p = 0; p < 2; ++p) {
        const int c   = tid + p * 1024;           // chunk 0..2047
        const int r   = c >> 7;                   // row 0..15 (128 chunks/row)
        const int col = (c & 127) * 4;            // f32 column
        async_copy_b128(xw + ((size_t)(bOff + r) * RNN_L + 0) * RNN_H + col,
                        &xbuf[0][r * RNN_H + col]);
    }
#endif

    for (int t = 0; t < RNN_L; ++t) {
        const int cur = t & 1;

#if USE_ASYNC_LDS
        // stage xw[t+1] into the other buffer while we do the matmul
        if (t + 1 < RNN_L) {
#pragma unroll
            for (int p = 0; p < 2; ++p) {
                const int c   = tid + p * 1024;
                const int r   = c >> 7;
                const int col = (c & 127) * 4;
                async_copy_b128(xw + ((size_t)(bOff + r) * RNN_L + (t + 1)) * RNN_H + col,
                                &xbuf[cur ^ 1][r * RNN_H + col]);
            }
        }
#endif

        v8f acc = {};
        // K-chunks 0..7: B from registers, A from LDS
#pragma unroll
        for (int c = 0; c < 8; ++c) {
            BF16x16 a = load_frag(aRow, c * 32, koff);
            acc = wmma_bf16(a, bReg[c], acc);
        }
        // K-chunks 8..15: B streamed (L2-resident), A from LDS
#pragma unroll 4
        for (int c = 8; c < 16; ++c) {
            BF16x16 b = load_frag(bRow, c * 32, koff);
            BF16x16 a = load_frag(aRow, c * 32, koff);
            acc = wmma_bf16(a, b, acc);
        }

#if USE_ASYNC_LDS
        // my 2 copies for xbuf[cur] were issued last step; async loads retire
        // in order, so allowing the 2 just-issued (t+1) copies to remain
        // pending is enough to guarantee xbuf[cur] is resident.
        if (t + 1 < RNN_L) __builtin_amdgcn_s_wait_asynccnt(2);
        else               __builtin_amdgcn_s_wait_asynccnt(0);
#endif
        __syncthreads();  // h reads done AND (async path) xbuf[cur] visible WG-wide

#pragma unroll
        for (int v = 0; v < 8; ++v) {
            const int bl = mHalf + v;          // local batch row 0..15
            const int bg = bOff + bl;          // global batch row
            const size_t idx = ((size_t)bg * RNN_L + t) * RNN_H + n;
#if USE_ASYNC_LDS
            const float xv = xbuf[cur][bl * RNN_H + n];
#else
            const float xv = xw[idx];
#endif
            const float hn = tanhf(xv + acc[v]);
            hsm[bl * RNN_H + n] = (__bf16)hn;
            if (seq_bf)  seq_bf[idx]  = (__bf16)hn;
            if (seq_f32) seq_f32[idx] = hn;
            if (t == RNN_L - 1) hN[(size_t)bg * RNN_H + n] = hn;
        }
        __syncthreads();  // hnew visible before next step's reads
    }
}

// ---------------------------------------------------------------------------
// Launcher. Inputs: x, h_0, WI0, BI0, WH0, BH0, WI1, BI1, WH1, BH1 (all f32).
// d_out = seq[32,1024,512] f32  ++  h_n[2,32,512] f32.
// ---------------------------------------------------------------------------
extern "C" void kernel_launch(void* const* d_in, const int* in_sizes, int n_in,
                              void* d_out, int out_size, void* d_ws, size_t ws_size,
                              hipStream_t stream) {
    (void)in_sizes; (void)n_in; (void)out_size; (void)ws_size;

    const float* x   = (const float*)d_in[0];
    const float* h_0 = (const float*)d_in[1];
    const float* WI0 = (const float*)d_in[2];
    const float* BI0 = (const float*)d_in[3];
    const float* WH0 = (const float*)d_in[4];
    const float* BH0 = (const float*)d_in[5];
    const float* WI1 = (const float*)d_in[6];
    const float* BI1 = (const float*)d_in[7];
    const float* WH1 = (const float*)d_in[8];
    const float* BH1 = (const float*)d_in[9];

    float* out = (float*)d_out;
    float* seq_out = out;                                        // [32,1024,512]
    float* hN0 = out + (size_t)RNN_B * RNN_L * RNN_H;            // h_n layer 0
    float* hN1 = hN0 + (size_t)RNN_B * RNN_H;                    // h_n layer 1

    // workspace layout
    char* ws = (char*)d_ws;
    __bf16* Abf   = (__bf16*)(ws);                               // 32 MB: x / seq bf16
    float*  XW    = (float*)(ws + (size_t)(32 << 20));           // 64 MB: xw f32
    __bf16* WIt0  = (__bf16*)(ws + (size_t)(96 << 20));          // 512 KB each
    __bf16* WHt0  = WIt0 + (size_t)RNN_H * RNN_H;
    __bf16* WIt1  = WHt0 + (size_t)RNN_H * RNN_H;
    __bf16* WHt1  = WIt1 + (size_t)RNN_H * RNN_H;
    float*  bias0 = (float*)(ws + (size_t)(98 << 20));
    float*  bias1 = bias0 + RNN_H;

    const int nX = RNN_B * RNN_L * RNN_H;      // 16,777,216
    const int nW = RNN_H * RNN_H;              // 262,144

    // --- prep: conversions / transposes / bias sums -------------------------
    cvt_f32_to_bf16<<<4096, 256, 0, stream>>>(x, Abf, nX);
    transpose_to_bf16<<<(nW + 255) / 256, 256, 0, stream>>>(WI0, WIt0, RNN_H, RNN_H);
    transpose_to_bf16<<<(nW + 255) / 256, 256, 0, stream>>>(WH0, WHt0, RNN_H, RNN_H);
    transpose_to_bf16<<<(nW + 255) / 256, 256, 0, stream>>>(WI1, WIt1, RNN_H, RNN_H);
    transpose_to_bf16<<<(nW + 255) / 256, 256, 0, stream>>>(WH1, WHt1, RNN_H, RNN_H);
    bias_sum<<<2, 256, 0, stream>>>(BI0, BH0, bias0, RNN_H);
    bias_sum<<<2, 256, 0, stream>>>(BI1, BH1, bias1, RNN_H);

    const dim3 gemmGrid(RNN_M / 128, RNN_H / 64);   // (256, 8)

    // --- layer 0 ------------------------------------------------------------
    gemm_xw_bf16<<<gemmGrid, 256, 0, stream>>>(Abf, WIt0, bias0, XW, RNN_M);
    // scan writes next-layer bf16 input back into Abf (x bf16 no longer needed)
    rnn_scan<<<2, 1024, 0, stream>>>(XW, WHt0, h_0, Abf, nullptr, hN0);

    // --- layer 1 ------------------------------------------------------------
    gemm_xw_bf16<<<gemmGrid, 256, 0, stream>>>(Abf, WIt1, bias1, XW, RNN_M);
    rnn_scan<<<2, 1024, 0, stream>>>(XW, WHt1, h_0 + (size_t)RNN_B * RNN_H,
                                     nullptr, seq_out, hN1);
}
